// GregTransformer_2680059592717
// MI455X (gfx1250) — matrix operand order repaired and checked by reading, not compiled, not run on hardware
//
#include <hip/hip_runtime.h>
#include <hip/hip_bf16.h>
#include <math.h>

typedef unsigned int   u32;
typedef unsigned short u16;

#define D_MODEL 1024
#define N_HEADS 16
#define HEAD_DIM 64
#define SEQ 2048
#define BATCH 2
#define FFN_DIM 4096
#define BS (BATCH * SEQ)   // 4096 rows

typedef __attribute__((ext_vector_type(16))) __bf16 v16bf;
typedef __attribute__((ext_vector_type(8)))  float  v8f;
typedef __attribute__((ext_vector_type(4)))  int    v4i;

union frag_t {
    v16bf v;
    u32   u[8];
};

// ---------------------------------------------------------------------------
// CDNA5 async global->LDS staging (ASYNCcnt path), with compile-safe fallback
// ---------------------------------------------------------------------------
#if defined(__HIP_DEVICE_COMPILE__) && defined(__gfx1250__)
#if __has_builtin(__builtin_amdgcn_global_load_async_to_lds_b128)
#define USE_ASYNC_LDS 1
#endif
#endif

#ifdef USE_ASYNC_LDS
typedef __attribute__((address_space(1))) v4i* gas_v4i_ptr;
typedef __attribute__((address_space(3))) v4i* las_v4i_ptr;
#endif

static __device__ __forceinline__ void stage16(const u16* g, u16* l) {
#ifdef USE_ASYNC_LDS
    __builtin_amdgcn_global_load_async_to_lds_b128(
        (gas_v4i_ptr)(void*)g, (las_v4i_ptr)(void*)l, 0, 0);
#else
    *(uint4*)l = *(const uint4*)g;
#endif
}

static __device__ __forceinline__ void async_stage_wait() {
#ifdef USE_ASYNC_LDS
#if __has_builtin(__builtin_amdgcn_s_wait_asynccnt)
    __builtin_amdgcn_s_wait_asynccnt(0);
#else
    asm volatile("s_wait_asynccnt 0x0" ::: "memory");
#endif
#endif
}

static __device__ __forceinline__ u16 f2bf(float f) {
    u32 u = __float_as_uint(f);
    u32 r = u + 0x7fffu + ((u >> 16) & 1u);   // round-to-nearest-even
    return (u16)(r >> 16);
}

static __device__ __forceinline__ v8f wmma_bf16(const frag_t& a, const frag_t& b, v8f c) {
    // D(16x16,f32) = A(16x32,bf16) * B(32x16,bf16) + C
    return __builtin_amdgcn_wmma_f32_16x16x32_bf16(
        /*neg_a=*/false, a.v, /*neg_b=*/false, b.v,
        /*c_mod=*/(short)0, c, /*reuse_a=*/false, /*reuse_b=*/false);
}

static __device__ __forceinline__ float gelu_exact(float x) {
    return 0.5f * x * (1.0f + erff(x * 0.70710678118654752f));
}

// ---------------------------------------------------------------------------
// fp32 -> bf16 conversion (vectorized by 4)
// ---------------------------------------------------------------------------
__global__ void cvt_bf16_kernel(const float* __restrict__ in, u16* __restrict__ out, int n4) {
    int i = blockIdx.x * blockDim.x + threadIdx.x;
    if (i >= n4) return;
    float4 f = *(const float4*)(in + (size_t)i * 4);
    uint2 p;
    p.x = (u32)f2bf(f.x) | ((u32)f2bf(f.y) << 16);
    p.y = (u32)f2bf(f.z) | ((u32)f2bf(f.w) << 16);
    *(uint2*)(out + (size_t)i * 4) = p;
}

// ---------------------------------------------------------------------------
// LayerNorm (one 256-thread block per row of D_MODEL=1024) -> bf16 output
// ---------------------------------------------------------------------------
__global__ __launch_bounds__(256) void ln_bf16_kernel(const float* __restrict__ x,
                                                      const float* __restrict__ g,
                                                      const float* __restrict__ beta,
                                                      u16* __restrict__ out) {
    int row = blockIdx.x;
    int t = threadIdx.x, lane = t & 31, wave = t >> 5;
    const float* xr = x + (size_t)row * D_MODEL;
    float4 xv = *(const float4*)(xr + t * 4);
    float s  = xv.x + xv.y + xv.z + xv.w;
    float sq = xv.x * xv.x + xv.y * xv.y + xv.z * xv.z + xv.w * xv.w;
    for (int o = 16; o > 0; o >>= 1) {
        s  += __shfl_down(s,  o, 32);
        sq += __shfl_down(sq, o, 32);
    }
    __shared__ float as_[8], aq_[8];
    __shared__ float mu_s, rstd_s;
    if (lane == 0) { as_[wave] = s; aq_[wave] = sq; }
    __syncthreads();
    if (t == 0) {
        float S = 0.f, Q = 0.f;
        for (int i = 0; i < 8; ++i) { S += as_[i]; Q += aq_[i]; }
        float m = S * (1.0f / D_MODEL);
        mu_s = m;
        rstd_s = rsqrtf(Q * (1.0f / D_MODEL) - m * m + 1e-5f);
    }
    __syncthreads();
    float m = mu_s, r = rstd_s;
    const float4 gv = *(const float4*)(g + t * 4);
    const float4 bv = *(const float4*)(beta + t * 4);
    float y0 = (xv.x - m) * r * gv.x + bv.x;
    float y1 = (xv.y - m) * r * gv.y + bv.y;
    float y2 = (xv.z - m) * r * gv.z + bv.z;
    float y3 = (xv.w - m) * r * gv.w + bv.w;
    uint2 p;
    p.x = (u32)f2bf(y0) | ((u32)f2bf(y1) << 16);
    p.y = (u32)f2bf(y2) | ((u32)f2bf(y3) << 16);
    *(uint2*)(out + (size_t)row * D_MODEL + t * 4) = p;
}

// ---------------------------------------------------------------------------
// Generic bf16 WMMA GEMM: C[M,N] = act(A[M,K] * B[K,N] + bias) (+ residual)
// block = 256 threads (8 waves); tile BM=128, BN=64, BK=64.
// Each wave owns a 16x64 strip: 4 C fragments, 8 WMMAs per staging round.
// ---------------------------------------------------------------------------
template <int ACT, bool RESID, bool OUTBF>
__global__ __launch_bounds__(256) void gemm_bf16_kernel(const u16* __restrict__ A,
                                                        const u16* __restrict__ Bm,
                                                        const float* __restrict__ bias,
                                                        const float* __restrict__ resid,
                                                        void* __restrict__ outp,
                                                        int M, int N, int K) {
    __shared__ u16 As[128][64];   // A tile, row-major (K contiguous)   16 KB
    __shared__ u16 Bt[64][64];    // B tile transposed: [n][k]           8 KB
    int t = threadIdx.x, lane = t & 31, wave = t >> 5;
    int tile_n = blockIdx.x * 64;
    int tile_m = blockIdx.y * 128;
    int lm   = lane & 15;
    int koff = (lane >= 16) ? 8 : 0;    // A-fragment hi-lane K offset
    int kof2 = (lane >= 16) ? 16 : 0;   // unused; B fragment also uses koff

    v8f acc[4] = {};

    for (int k0 = 0; k0 < K; k0 += 64) {
        // stage A: 128x64 bf16 = 1024 x (8xb16) chunks, 4 per thread (async DMA)
        #pragma unroll
        for (int j = 0; j < 4; ++j) {
            int c = t + 256 * j;
            int row = c >> 3, off = (c & 7) * 8;
            stage16(A + (size_t)(tile_m + row) * K + k0 + off, &As[row][off]);
        }
        // stage B transposed: 64x64 source, load uint (2 n's), scatter to [n][k]
        #pragma unroll
        for (int j = 0; j < 8; ++j) {
            int c = t + 256 * j;
            int kk = c >> 5, n2 = (c & 31) * 2;
            u32 w = *(const u32*)(Bm + (size_t)(k0 + kk) * N + tile_n + n2);
            Bt[n2][kk]     = (u16)(w & 0xffffu);
            Bt[n2 + 1][kk] = (u16)(w >> 16);
        }
        if (k0 + 64 < K) {
            __builtin_prefetch(A + (size_t)(tile_m + (t >> 3)) * K + k0 + 64, 0, 0);
            __builtin_prefetch(Bm + (size_t)(k0 + 64 + (t >> 5)) * N + tile_n, 0, 0);
        }
        async_stage_wait();
        __syncthreads();

        // A fragments (2 K-chunks of 32), ISA layout (K pairs per VGPR, hi lanes +8)
        frag_t af[2];
        #pragma unroll
        for (int kc = 0; kc < 2; ++kc)
            #pragma unroll
            for (int v = 0; v < 8; ++v) {
                int kk = kc * 32 + ((v < 4) ? 2 * v : 2 * (v - 4) + 16) + koff;
                af[kc].u[v] = *(const u32*)&As[wave * 16 + lm][kk];
            }
        #pragma unroll
        for (int kc = 0; kc < 2; ++kc)
            #pragma unroll
            for (int nt = 0; nt < 4; ++nt) {
                frag_t bf;
                #pragma unroll
                for (int v = 0; v < 8; ++v)
                    bf.u[v] = *(const u32*)&Bt[nt * 16 + lm][kc * 32 + ((v < 4) ? 2 * v : 2 * (v - 4) + 16) + koff];
                acc[nt] = wmma_bf16(af[kc], bf, acc[nt]);
            }
        __syncthreads();
    }

    // epilogue: C layout -> lane (M = r + 8*hi, N = lane%16)
    #pragma unroll
    for (int nt = 0; nt < 4; ++nt) {
        int n = tile_n + nt * 16 + lm;
        float bv = bias ? bias[n] : 0.0f;
        #pragma unroll
        for (int r = 0; r < 8; ++r) {
            int m = tile_m + wave * 16 + r + ((lane >= 16) ? 8 : 0);
            float v = acc[nt][r] + bv;
            if (ACT == 1) v = gelu_exact(v);
            if (RESID)    v += resid[(size_t)m * N + n];
            if (OUTBF) ((u16*)outp)[(size_t)m * N + n] = f2bf(v);
            else       ((float*)outp)[(size_t)m * N + n] = v;
        }
    }
}

// ---------------------------------------------------------------------------
// Fused causal ReLU-kernel attention: ctx = (relu(Q K^T / 8) .* tril) V
// One block (128 thr = 4 waves) per (b, h, 64-row q tile); 64-key blocks.
// Each wave owns 16 query rows; Q fragments live in registers across the loop.
// 16 WMMAs per wave per staging round (8 QK^T + 8 PV).
// ---------------------------------------------------------------------------
__global__ __launch_bounds__(128) void attn_relu_causal_kernel(const u16* __restrict__ qb,
                                                               const u16* __restrict__ kptr,
                                                               const u16* __restrict__ vptr,
                                                               u16* __restrict__ ctxb) {
    int bid = blockIdx.x;
    int qt = bid & (SEQ / 64 - 1);
    int tmp = bid / (SEQ / 64);
    int h = tmp & (N_HEADS - 1);
    int b = tmp / N_HEADS;
    int t = threadIdx.x, lane = t & 31, wave = t >> 5;
    int q0 = qt * 64;
    int lm   = lane & 15;
    int koff = (lane >= 16) ? 8 : 0;
    int kof2 = (lane >= 16) ? 16 : 0;

    __shared__ u16 Kt[64][64];      // [key][dh]                          8 KB
    __shared__ u16 Vtt[64][66];     // [dh][key] (transposed, padded)  ~8.25 KB
    __shared__ u16 Pt[4][16][64];   // per-wave P tile [m][key_local]     8 KB

    // Q fragments: 16 rows x 64 (Dh), 2 K-chunks of 32 -> registers, loaded once
    frag_t qf[2];
    const u16* qbase = qb + ((size_t)(b * SEQ) + q0 + wave * 16 + lm) * D_MODEL + h * HEAD_DIM;
    #pragma unroll
    for (int f = 0; f < 2; ++f)
        #pragma unroll
        for (int v = 0; v < 8; ++v) {
            int kk = f * 32 + ((v < 4) ? 2 * v : 2 * (v - 4) + 16) + koff;
            qf[f].u[v] = *(const u32*)(qbase + kk);
        }

    v8f ctx[4] = {};

    for (int kbi = 0; kbi <= qt; ++kbi) {     // causal: 64-key blocks up to q tile
        int kb0 = kbi * 64;
        // stage K tile (64x64): 512 x (8xb16) chunks, 4 per thread (async DMA)
        #pragma unroll
        for (int j = 0; j < 4; ++j) {
            int c = t + 128 * j;
            int row = c >> 3, off = (c & 7) * 8;
            stage16(kptr + ((size_t)(b * SEQ) + kb0 + row) * D_MODEL + h * HEAD_DIM + off,
                    &Kt[row][off]);
        }
        // stage V transposed (dh-major): uint4 global loads, 8 scalar LDS scatters
        #pragma unroll
        for (int j = 0; j < 4; ++j) {
            int c = t + 128 * j;
            int row = c >> 3, dh0 = (c & 7) * 8;
            uint4 w = *(const uint4*)(vptr + ((size_t)(b * SEQ) + kb0 + row) * D_MODEL
                                      + h * HEAD_DIM + dh0);
            const u16* parts = (const u16*)&w;
            #pragma unroll
            for (int i = 0; i < 8; ++i)
                Vtt[dh0 + i][row] = parts[i];
        }
        if (kbi < qt) {
            __builtin_prefetch(kptr + ((size_t)(b * SEQ) + kb0 + 64 + (t >> 1)) * D_MODEL
                               + h * HEAD_DIM, 0, 0);
            __builtin_prefetch(vptr + ((size_t)(b * SEQ) + kb0 + 64 + (t >> 1)) * D_MODEL
                               + h * HEAD_DIM, 0, 0);
        }
        async_stage_wait();
        __syncthreads();

        // scores: 16x64 per wave = 4 N tiles x 2 K chunks of WMMA
        v8f sc[4] = {};
        #pragma unroll
        for (int nt = 0; nt < 4; ++nt) {
            int key = nt * 16 + lm;
            #pragma unroll
            for (int kc = 0; kc < 2; ++kc) {
                frag_t bf;   // B = K^T : K-dim = dh, N = key
                #pragma unroll
                for (int v = 0; v < 8; ++v)
                    bf.u[v] = *(const u32*)&Kt[key][kc * 32 + ((v < 4) ? 2 * v : 2 * (v - 4) + 16) + koff];
                sc[nt] = wmma_bf16(qf[kc], bf, sc[nt]);
            }
        }
        // scale + ReLU + causal mask in C-fragment registers, write P to LDS (bf16)
        #pragma unroll
        for (int nt = 0; nt < 4; ++nt) {
            int keyl = nt * 16 + lm;
            int key = kb0 + keyl;
            #pragma unroll
            for (int r = 0; r < 8; ++r) {
                int m = r + ((lane >= 16) ? 8 : 0);
                int qg = q0 + wave * 16 + m;
                float s = sc[nt][r] * 0.125f;
                s = (key <= qg && s > 0.0f) ? s : 0.0f;
                Pt[wave][m][keyl] = f2bf(s);
            }
        }
        __syncthreads();

        // ctx += P(16x64) * V(64x64): re-fragment P as A operand, 8 WMMAs
        frag_t pf[2];
        #pragma unroll
        for (int kc = 0; kc < 2; ++kc)
            #pragma unroll
            for (int v = 0; v < 8; ++v) {
                int kk = kc * 32 + ((v < 4) ? 2 * v : 2 * (v - 4) + 16) + koff;
                pf[kc].u[v] = *(const u32*)&Pt[wave][lm][kk];
            }
        #pragma unroll
        for (int kc = 0; kc < 2; ++kc)
            #pragma unroll
            for (int nt = 0; nt < 4; ++nt) {
                frag_t vf;   // B = V : K-dim = key, N = dh
                int dh = nt * 16 + lm;
                #pragma unroll
                for (int v = 0; v < 8; ++v)
                    vf.u[v] = *(const u32*)&Vtt[dh][kc * 32 + ((v < 4) ? 2 * v : 2 * (v - 4) + 16) + koff];
                ctx[nt] = wmma_bf16(pf[kc], vf, ctx[nt]);
            }
        __syncthreads();
    }

    // write ctx (bf16) to [B*S, D] at column block h*64
    u16* cb = ctxb + ((size_t)(b * SEQ) + q0 + wave * 16) * D_MODEL + h * HEAD_DIM;
    #pragma unroll
    for (int nt = 0; nt < 4; ++nt)
        #pragma unroll
        for (int r = 0; r < 8; ++r) {
            int m = r + ((lane >= 16) ? 8 : 0);
            cb[(size_t)m * D_MODEL + nt * 16 + lm] = f2bf(ctx[nt][r]);
        }
}

// ---------------------------------------------------------------------------
// Launch sequence
// ---------------------------------------------------------------------------
extern "C" void kernel_launch(void* const* d_in, const int* in_sizes, int n_in,
                              void* d_out, int out_size, void* d_ws, size_t ws_size,
                              hipStream_t stream) {
    const float* x     = (const float*)d_in[0];
    const float* wq    = (const float*)d_in[1];
    const float* bq    = (const float*)d_in[2];
    const float* wk    = (const float*)d_in[3];
    const float* bk    = (const float*)d_in[4];
    const float* wv    = (const float*)d_in[5];
    const float* bv    = (const float*)d_in[6];
    const float* wo    = (const float*)d_in[7];
    const float* bo    = (const float*)d_in[8];
    const float* g1    = (const float*)d_in[9];
    const float* beta1 = (const float*)d_in[10];
    const float* g2    = (const float*)d_in[11];
    const float* beta2 = (const float*)d_in[12];
    const float* w1    = (const float*)d_in[13];
    const float* b1    = (const float*)d_in[14];
    const float* w2    = (const float*)d_in[15];
    const float* b2    = (const float*)d_in[16];

    const size_t MB = 1ull << 20;
    char* base = (char*)d_ws;
    u16*   wqb  = (u16*)(base + 0 * MB);    // 2 MB
    u16*   wkb  = (u16*)(base + 2 * MB);
    u16*   wvb  = (u16*)(base + 4 * MB);
    u16*   wob  = (u16*)(base + 6 * MB);
    u16*   w1b  = (u16*)(base + 8 * MB);    // 8 MB
    u16*   w2b  = (u16*)(base + 16 * MB);   // 8 MB
    float* x2   = (float*)(base + 24 * MB); // 16 MB
    u16*   hbf  = (u16*)(base + 40 * MB);   // 8 MB (reused for LN2 output)
    u16*   qb   = (u16*)(base + 48 * MB);   // 8 MB
    u16*   kbp  = (u16*)(base + 56 * MB);   // 8 MB
    u16*   vbp  = (u16*)(base + 64 * MB);   // 8 MB
    u16*   ctxb = (u16*)(base + 72 * MB);   // 8 MB  -> total 80 MB
    u16*   f1b  = (u16*)(base + 48 * MB);   // 32 MB overlay (q/k/v/ctx dead by then)

    // 1) weights -> bf16
    {
        int n4 = D_MODEL * D_MODEL / 4;
        int blocks = (n4 + 255) / 256;
        cvt_bf16_kernel<<<blocks, 256, 0, stream>>>(wq, wqb, n4);
        cvt_bf16_kernel<<<blocks, 256, 0, stream>>>(wk, wkb, n4);
        cvt_bf16_kernel<<<blocks, 256, 0, stream>>>(wv, wvb, n4);
        cvt_bf16_kernel<<<blocks, 256, 0, stream>>>(wo, wob, n4);
        int n4f = D_MODEL * FFN_DIM / 4;
        int blocksf = (n4f + 255) / 256;
        cvt_bf16_kernel<<<blocksf, 256, 0, stream>>>(w1, w1b, n4f);
        cvt_bf16_kernel<<<blocksf, 256, 0, stream>>>(w2, w2b, n4f);
    }

    // 2) LN1 -> hbf (bf16)
    ln_bf16_kernel<<<BS, 256, 0, stream>>>(x, g1, beta1, hbf);

    // 3) QKV projections (bf16 WMMA, bias fused, bf16 out)
    {
        dim3 grid(D_MODEL / 64, BS / 128);
        gemm_bf16_kernel<0, false, true><<<grid, 256, 0, stream>>>(hbf, wqb, bq, nullptr, qb,  BS, D_MODEL, D_MODEL);
        gemm_bf16_kernel<0, false, true><<<grid, 256, 0, stream>>>(hbf, wkb, bk, nullptr, kbp, BS, D_MODEL, D_MODEL);
        gemm_bf16_kernel<0, false, true><<<grid, 256, 0, stream>>>(hbf, wvb, bv, nullptr, vbp, BS, D_MODEL, D_MODEL);
    }

    // 4) fused causal ReLU attention -> ctxb (bf16)
    attn_relu_causal_kernel<<<BATCH * N_HEADS * (SEQ / 64), 128, 0, stream>>>(qb, kbp, vbp, ctxb);

    // 5) output projection + residual -> x2 (fp32)
    {
        dim3 grid(D_MODEL / 64, BS / 128);
        gemm_bf16_kernel<0, true, false><<<grid, 256, 0, stream>>>(ctxb, wob, bo, x, x2, BS, D_MODEL, D_MODEL);
    }

    // 6) LN2 -> hbf (bf16, reuse)
    ln_bf16_kernel<<<BS, 256, 0, stream>>>(x2, g2, beta2, hbf);

    // 7) FFN up + exact GELU -> f1b (bf16)
    {
        dim3 grid(FFN_DIM / 64, BS / 128);
        gemm_bf16_kernel<1, false, true><<<grid, 256, 0, stream>>>(hbf, w1b, b1, nullptr, f1b, BS, FFN_DIM, D_MODEL);
    }

    // 8) FFN down + residual -> d_out (fp32)
    {
        dim3 grid(D_MODEL / 64, BS / 128);
        gemm_bf16_kernel<0, true, false><<<grid, 256, 0, stream>>>(f1b, w2b, b2, x2, (float*)d_out, BS, D_MODEL, FFN_DIM);
    }
}
